// DAM_Module_26542897889560
// MI455X (gfx1250) — matrix operand order repaired and checked by reading, not compiled, not run on hardware
//
#include <hip/hip_runtime.h>
#include <hip/hip_bf16.h>
#include <math.h>

// ---------------------------------------------------------------------------
// CDNA5 (gfx1250) wave32 WMMA: D(16x16,f32) = A(16x32,f16) x B(32x16,f16) + C
// Operand VGPR layouts used in this file (A and B share one k order):
//   A: lane holds row m = lane&15; element e -> k = ((lane>>4)<<4) + e
//   B: lane holds col n = lane&15; element e -> k = (lane&16) + e
//   C: element r -> row = r + ((lane>>4)<<3), col = lane&15
// LDS tiles stored "consumer-linear": element (k,n) at ((k&16)|n)*16+(k&15),
// so a consuming lane reads its 16 halfs contiguously (2 x ds_load_b128).
// ---------------------------------------------------------------------------

typedef __attribute__((ext_vector_type(16))) _Float16 v16h;
typedef __attribute__((ext_vector_type(8)))  float    v8f;

__device__ __forceinline__ int a_k_of(int lane, int e) {
  return ((lane >> 4) << 4) + e;
}
__device__ __forceinline__ v8f wmma_f16(v16h a, v16h b, v8f c) {
  return __builtin_amdgcn_wmma_f32_16x16x32_f16(false, a, false, b, (short)0, c,
                                                false, false);
}
__device__ __forceinline__ v8f zero8() {
  v8f z = {0.f, 0.f, 0.f, 0.f, 0.f, 0.f, 0.f, 0.f};
  return z;
}

#define NB   4
#define CIN  256
#define COUT 128
#define HW   4096   // 64*64
#define CQK  16
#define KCONV 2304  // 9*256

// ---------------------------------------------------------------------------
// prep kernels
// ---------------------------------------------------------------------------
__global__ void dam_cast_f32_f16(const float* __restrict__ src,
                                 _Float16* __restrict__ dst, int n) {
  int i = blockIdx.x * blockDim.x + threadIdx.x;
  if (i < n) dst[i] = (_Float16)src[i];
}

// conv_w (Cout,Cin,3,3) -> wA[co][tap*256+ci], tap = kh*3+kw, as f16
__global__ void dam_pack_conv_w(const float* __restrict__ w,
                                _Float16* __restrict__ wA, int total) {
  int i = blockIdx.x * blockDim.x + threadIdx.x;
  if (i >= total) return;
  int co  = i / KCONV;
  int r   = i - co * KCONV;
  int tap = r >> 8;
  int ci  = r & 255;
  int kh  = tap / 3, kw = tap - 3 * kh;
  wA[i] = (_Float16)w[((size_t)(co * CIN + ci) * 3 + kh) * 3 + kw];
}

// ---------------------------------------------------------------------------
// conv3x3 as implicit GEMM. Block = (b, 32-pixel macro tile); 8 waves cover
// 128 co, each wave keeps 2 accumulators (2 n-subtiles) reusing one A tile.
// Shifted-x B chunk (32ci x 32px) double-buffered in LDS; staging loads for
// chunk k+1 (branchless clamped) overlap the WMMAs of chunk k; one barrier
// per chunk. K = 9 taps x 8 channel slabs = 72 chunks of 32.
// ---------------------------------------------------------------------------
__global__ __launch_bounds__(256) void dam_conv_gemm(
    const _Float16* __restrict__ wA, const _Float16* __restrict__ xh,
    float* __restrict__ y) {
  __shared__ _Float16 Bs[2][1024];   // [buf][subtile(2) x consumer-linear 512]
  int b   = blockIdx.x >> 7;         // 128 blocks per batch
  int n0  = (blockIdx.x & 127) << 5; // 32-pixel macro tile
  int tid  = threadIdx.x;
  int wave = tid >> 5, lane = tid & 31;
  int m0   = wave << 4;
  int mrow = m0 + (lane & 15);

  // staging geometry: thread -> pixel sn (0..31), channels sk+8j (j=0..3)
  int sn  = tid & 31;
  int sk  = tid >> 5;                // 0..7
  int sng = n0 + sn;
  int sh = sng >> 6, sw = sng & 63;
  int snt = sn >> 4;                 // n-subtile
  int snn = sn & 15;

  v8f acc0 = zero8(), acc1 = zero8();
  const _Float16* wrow_base = wA + (size_t)mrow * KCONV;
  _Float16 sv[4];

  // stage chunk 0 into buffer 0
  {
    int hs = sh - 1, wsx = sw - 1;   // tap 0: dh=dw=-1
    bool inb = ((unsigned)hs < 64u) && ((unsigned)wsx < 64u);
    int pix = inb ? (hs * 64 + wsx) : 0;
    const _Float16* xp = xh + ((size_t)b * CIN + sk) * HW + pix;
#pragma unroll
    for (int j = 0; j < 4; ++j) {
      _Float16 v = xp[(size_t)(8 * j) * HW];
      sv[j] = inb ? v : (_Float16)0.f;
    }
#pragma unroll
    for (int j = 0; j < 4; ++j) {
      int k = sk + 8 * j;
      Bs[0][snt * 512 + (((k & 16) | snn) << 4) + (k & 15)] = sv[j];
    }
  }

  for (int kc = 0; kc < 72; ++kc) {
    int cur = kc & 1, nxt = cur ^ 1;
    __syncthreads();
    // prefetch chunk kc+1 into registers (overlaps WMMAs below)
    {
      int kn = kc + 1;
      int tap = kn >> 3, ci0 = (kn & 7) << 5;
      int dh = tap / 3 - 1, dw = tap % 3 - 1;
      int hs = sh + dh, wsx = sw + dw;
      bool inb = (tap < 9) && ((unsigned)hs < 64u) && ((unsigned)wsx < 64u);
      int pix = inb ? (hs * 64 + wsx) : 0;
      const _Float16* xp = xh + ((size_t)b * CIN + ci0 + sk) * HW + pix;
#pragma unroll
      for (int j = 0; j < 4; ++j) {
        _Float16 v = xp[(size_t)(8 * j) * HW];
        sv[j] = inb ? v : (_Float16)0.f;
      }
    }
    // consume current chunk: one A tile feeds two WMMAs
    {
      int tap = kc >> 3, cc = kc & 7;
      const _Float16* wr = wrow_base + tap * 256 + (cc << 5);
      v16h A, B0, B1;
#pragma unroll
      for (int e = 0; e < 16; ++e) A[e] = wr[a_k_of(lane, e)];
#pragma unroll
      for (int e = 0; e < 16; ++e) B0[e] = Bs[cur][(lane << 4) + e];
#pragma unroll
      for (int e = 0; e < 16; ++e) B1[e] = Bs[cur][512 + (lane << 4) + e];
      acc0 = wmma_f16(A, B0, acc0);
      acc1 = wmma_f16(A, B1, acc1);
    }
    // commit staged registers for next chunk
#pragma unroll
    for (int j = 0; j < 4; ++j) {
      int k = sk + 8 * j;
      Bs[nxt][snt * 512 + (((k & 16) | snn) << 4) + (k & 15)] = sv[j];
    }
  }
#pragma unroll
  for (int r = 0; r < 8; ++r) {
    int co = m0 + r + ((lane >> 4) << 3);
    size_t base = ((size_t)b * COUT + co) * HW + n0 + (lane & 15);
    y[base]      = acc0[r];
    y[base + 16] = acc1[r];
  }
}

// ---------------------------------------------------------------------------
// BatchNorm stats (training mode) -> per-channel scale/shift
// ---------------------------------------------------------------------------
__global__ __launch_bounds__(256) void dam_bn_stats(
    const float* __restrict__ y, const float* __restrict__ gamma,
    const float* __restrict__ beta, float* __restrict__ scale,
    float* __restrict__ shift) {
  __shared__ float s1[256], s2[256];
  int co = blockIdx.x, tid = threadIdx.x;
  float a = 0.f, q = 0.f;
  for (int i = tid; i < NB * HW; i += 256) {
    int b = i >> 12, n = i & (HW - 1);
    float v = y[((size_t)b * COUT + co) * HW + n];
    a += v;
    q += v * v;
  }
  s1[tid] = a; s2[tid] = q;
  __syncthreads();
  for (int s = 128; s > 0; s >>= 1) {
    if (tid < s) { s1[tid] += s1[tid + s]; s2[tid] += s2[tid + s]; }
    __syncthreads();
  }
  if (tid == 0) {
    const float inv = 1.f / (float)(NB * HW);
    float mean = s1[0] * inv;
    float var  = s2[0] * inv - mean * mean;
    float sc   = gamma[co] * rsqrtf(var + 1e-5f);
    scale[co] = sc;
    shift[co] = beta[co] - mean * sc;
  }
}

__global__ void dam_bn_relu(const float* __restrict__ y,
                            const float* __restrict__ scale,
                            const float* __restrict__ shift,
                            float* __restrict__ feat,
                            _Float16* __restrict__ feath, int n) {
  int i = blockIdx.x * blockDim.x + threadIdx.x;
  if (i >= n) return;
  int c = (i >> 12) & (COUT - 1);
  float v = fmaf(y[i], scale[c], shift[c]);
  v = v > 0.f ? v : 0.f;
  feat[i]  = v;
  feath[i] = (_Float16)v;
}

// ---------------------------------------------------------------------------
// projection (K=128). transposed=1 writes [b][n][M] (q/k, M=16) for contiguous
// PAM-energy operands; transposed=0 writes [b][o][n] (v).
// ---------------------------------------------------------------------------
__global__ __launch_bounds__(256) void dam_proj_gemm(
    const _Float16* __restrict__ Wh, const float* __restrict__ bias,
    const _Float16* __restrict__ fh, _Float16* __restrict__ outh,
    int Mtiles, int transposed) {
  int wid  = blockIdx.x * 8 + (threadIdx.x >> 5);
  int lane = threadIdx.x & 31;
  int wpb = Mtiles << 8;
  int b = wid / wpb;
  int t = wid - b * wpb;
  int m0 = (t >> 8) << 4;
  int n0 = (t & 255) << 4;
  int M  = Mtiles << 4;

  v8f acc = zero8();
  const _Float16* wr = Wh + (size_t)(m0 + (lane & 15)) * COUT;
  for (int kc = 0; kc < 4; ++kc) {
    v16h A, Bv;
#pragma unroll
    for (int e = 0; e < 16; ++e) A[e] = wr[kc * 32 + a_k_of(lane, e)];
#pragma unroll
    for (int e = 0; e < 16; ++e)
      Bv[e] = fh[((size_t)b * COUT + kc * 32 + (lane & 16) + e) * HW + n0 +
                 (lane & 15)];
    acc = wmma_f16(A, Bv, acc);
  }
#pragma unroll
  for (int r = 0; r < 8; ++r) {
    int o = m0 + r + ((lane >> 4) << 3);
    int n = n0 + (lane & 15);
    float v = acc[r] + bias[o];
    if (transposed)
      outh[((size_t)b * HW + n) * CQK + o] = (_Float16)v;
    else
      outh[((size_t)b * M + o) * HW + n] = (_Float16)v;
  }
}

// ---------------------------------------------------------------------------
// CAM energy: energy[b][c][d] = sum_n f[c,n] f[d,n]  (M=N=128, K=4096)
// ---------------------------------------------------------------------------
__global__ __launch_bounds__(256) void dam_cam_energy(
    const _Float16* __restrict__ fh, float* __restrict__ energy) {
  int wid  = blockIdx.x * 8 + (threadIdx.x >> 5);
  int lane = threadIdx.x & 31;
  int b  = wid >> 6;
  int t  = wid & 63;
  int m0 = (t >> 3) << 4;
  int n0 = (t & 7) << 4;

  v8f acc = zero8();
  const _Float16* arow = fh + ((size_t)b * COUT + m0 + (lane & 15)) * HW;
  const _Float16* brow = fh + ((size_t)b * COUT + n0 + (lane & 15)) * HW;
  for (int kc = 0; kc < 128; ++kc) {
    v16h A, Bv;
#pragma unroll
    for (int e = 0; e < 16; ++e) A[e] = arow[kc * 32 + a_k_of(lane, e)];
#pragma unroll
    for (int e = 0; e < 16; ++e) Bv[e] = brow[kc * 32 + (lane & 16) + e];
    acc = wmma_f16(A, Bv, acc);
  }
#pragma unroll
  for (int r = 0; r < 8; ++r) {
    int c = m0 + r + ((lane >> 4) << 3);
    int d = n0 + (lane & 15);
    energy[((size_t)b * COUT + c) * COUT + d] = acc[r];
  }
}

// attn[c,d] = tanh(rowmax_c - energy[c,d])
__global__ void dam_cam_attn(const float* __restrict__ energy,
                             _Float16* __restrict__ attnh) {
  __shared__ float sm[COUT];
  int bc = blockIdx.x;
  int d  = threadIdx.x;
  float e = energy[(size_t)bc * COUT + d];
  sm[d] = e;
  __syncthreads();
  for (int s = 64; s > 0; s >>= 1) {
    if (d < s) sm[d] = fmaxf(sm[d], sm[d + s]);
    __syncthreads();
  }
  attnh[(size_t)bc * COUT + d] = (_Float16)tanhf(sm[0] - e);
}

// out = gamma_ca * (attn @ f) + 3*feat   (M=128, K=128, N=4096)
__global__ __launch_bounds__(256) void dam_cam_out(
    const _Float16* __restrict__ attnh, const _Float16* __restrict__ fh,
    const float* __restrict__ feat, const float* __restrict__ gca_p,
    float* __restrict__ out) {
  int wid  = blockIdx.x * 8 + (threadIdx.x >> 5);
  int lane = threadIdx.x & 31;
  int b  = wid >> 11;
  int t  = wid & 2047;
  int m0 = (t >> 8) << 4;
  int n0 = (t & 255) << 4;
  float gca = gca_p[0];

  v8f acc = zero8();
  const _Float16* arow = attnh + ((size_t)b * COUT + m0 + (lane & 15)) * COUT;
  for (int kc = 0; kc < 4; ++kc) {
    v16h A, Bv;
#pragma unroll
    for (int e = 0; e < 16; ++e) A[e] = arow[kc * 32 + a_k_of(lane, e)];
#pragma unroll
    for (int e = 0; e < 16; ++e)
      Bv[e] = fh[((size_t)b * COUT + kc * 32 + (lane & 16) + e) * HW + n0 +
                 (lane & 15)];
    acc = wmma_f16(A, Bv, acc);
  }
#pragma unroll
  for (int r = 0; r < 8; ++r) {
    int c = m0 + r + ((lane >> 4) << 3);
    int n = n0 + (lane & 15);
    size_t idx = ((size_t)b * COUT + c) * HW + n;
    out[idx] = gca * acc[r] + 3.f * feat[idx];
  }
}

// ---------------------------------------------------------------------------
// PAM, flash-style. Block = (b, 32-wide m macro tile); 8 waves cover 128 c
// rows with 2 accumulators each (one v A-tile feeds 2 WMMAs). Waves 0..3
// produce the four 16x16 tanh(q^T k) half-tiles for chunk nc+1 (2 m-subtiles
// x 2 n-halves) into the alternate LDS buffer while all 8 waves consume
// chunk nc: one barrier per chunk, production overlapped. q/k are stored
// transposed [b][n][16] so energy operands are contiguous b128 loads; the
// q^T A operand is loop-invariant. Attention (4x4096x4096) never hits HBM.
// ---------------------------------------------------------------------------
__global__ __launch_bounds__(256) void dam_pam_fused(
    const _Float16* __restrict__ qT, const _Float16* __restrict__ kT,
    const _Float16* __restrict__ vh, const float* __restrict__ gpa_p,
    float* __restrict__ out) {
  __shared__ _Float16 T[2][2][512];  // [buf][m-subtile][consumer-linear]
  int b    = blockIdx.x >> 7;
  int m0   = (blockIdx.x & 127) << 5;   // 32 m values
  int wave = threadIdx.x >> 5;
  int lane = threadIdx.x & 31;
  int c0   = wave << 4;
  float gpa = gpa_p[0];
  bool lo = lane < 16;
  int mt = (wave >> 1) & 1;  // producer m-subtile (waves 0..3)
  int hf = wave & 1;         // producer n-half

  // loop-invariant q^T A operand (K = 16 zero-padded to 32); waves 0..3
  v16h Aq;
  if (wave < 4) {
    const _Float16* qrow =
        qT + ((size_t)b * HW + m0 + (mt << 4) + (lane & 15)) * CQK;
#pragma unroll
    for (int e = 0; e < 16; ++e) {
      int kk = a_k_of(lane, e);
      _Float16 v = qrow[kk & 15];          // always-valid address
      Aq[e] = (kk < CQK) ? v : (_Float16)0.f;
    }
  }

  v8f acc0 = zero8(), acc1 = zero8();
  const _Float16* vrow = vh + ((size_t)b * COUT + c0 + (lane & 15)) * HW;

  // produce chunk 0 into buffer 0
  if (wave < 4) {
    int nb = hf << 4;
    const _Float16* krow = kT + ((size_t)b * HW + nb + (lane & 15)) * CQK;
    v16h Bk;
#pragma unroll
    for (int e = 0; e < 16; ++e) {
      _Float16 v = krow[e];
      Bk[e] = lo ? v : (_Float16)0.f;
    }
    v8f ez = wmma_f16(Aq, Bk, zero8());
#pragma unroll
    for (int r = 0; r < 8; ++r) {
      int m = r + ((lane >> 4) << 3);
      T[0][mt][(((hf << 4) + m) << 4) + (lane & 15)] = (_Float16)tanhf(ez[r]);
    }
  }

  for (int nc = 0; nc < 128; ++nc) {
    int cur = nc & 1, nxt = cur ^ 1;
    __syncthreads();
    if (wave < 4 && nc + 1 < 128) {
      // produce chunk nc+1 into the alternate buffer (overlaps WMMAs below)
      int nb = ((nc + 1) << 5) + (hf << 4);
      const _Float16* krow = kT + ((size_t)b * HW + nb + (lane & 15)) * CQK;
      v16h Bk;
#pragma unroll
      for (int e = 0; e < 16; ++e) {
        _Float16 v = krow[e];
        Bk[e] = lo ? v : (_Float16)0.f;
      }
      v8f ez = wmma_f16(Aq, Bk, zero8());
#pragma unroll
      for (int r = 0; r < 8; ++r) {
        int m = r + ((lane >> 4) << 3);
        T[nxt][mt][(((hf << 4) + m) << 4) + (lane & 15)] =
            (_Float16)tanhf(ez[r]);
      }
    }
    int n0 = nc << 5;
    v16h Av, B0, B1;
#pragma unroll
    for (int e = 0; e < 16; ++e) Av[e] = vrow[n0 + a_k_of(lane, e)];
#pragma unroll
    for (int e = 0; e < 16; ++e) B0[e] = T[cur][0][(lane << 4) + e];
#pragma unroll
    for (int e = 0; e < 16; ++e) B1[e] = T[cur][1][(lane << 4) + e];
    acc0 = wmma_f16(Av, B0, acc0);
    acc1 = wmma_f16(Av, B1, acc1);
  }
#pragma unroll
  for (int r = 0; r < 8; ++r) {
    int c = c0 + r + ((lane >> 4) << 3);
    size_t base = ((size_t)b * COUT + c) * HW + m0 + (lane & 15);
    out[base]      += gpa * acc0[r];
    out[base + 16] += gpa * acc1[r];
  }
}

// ---------------------------------------------------------------------------
// host launcher
// ---------------------------------------------------------------------------
extern "C" void kernel_launch(void* const* d_in, const int* in_sizes, int n_in,
                              void* d_out, int out_size, void* d_ws,
                              size_t ws_size, hipStream_t stream) {
  const float* x        = (const float*)d_in[0];
  const float* conv_w   = (const float*)d_in[1];
  const float* bn_gamma = (const float*)d_in[2];
  const float* bn_beta  = (const float*)d_in[3];
  const float* q_w      = (const float*)d_in[4];
  const float* q_b      = (const float*)d_in[5];
  const float* k_w      = (const float*)d_in[6];
  const float* k_b      = (const float*)d_in[7];
  const float* v_w      = (const float*)d_in[8];
  const float* v_b      = (const float*)d_in[9];
  const float* gca      = (const float*)d_in[10];
  const float* gpa      = (const float*)d_in[11];
  float* out = (float*)d_out;

  char* ws = (char*)d_ws;
  size_t off = 0;
  auto alloc = [&](size_t bytes) -> void* {
    void* p = ws + off;
    off = (off + bytes + 255) & ~(size_t)255;
    return p;
  };

  _Float16* xh    = (_Float16*)alloc((size_t)NB * CIN * HW * 2);
  _Float16* wA    = (_Float16*)alloc((size_t)COUT * KCONV * 2);
  float*    y     = (float*)   alloc((size_t)NB * COUT * HW * 4);
  float*    feat  = (float*)   alloc((size_t)NB * COUT * HW * 4);
  _Float16* feath = (_Float16*)alloc((size_t)NB * COUT * HW * 2);
  float*    scale = (float*)   alloc(COUT * 4);
  float*    shift = (float*)   alloc(COUT * 4);
  _Float16* qwh   = (_Float16*)alloc(CQK * COUT * 2);
  _Float16* kwh   = (_Float16*)alloc(CQK * COUT * 2);
  _Float16* vwh   = (_Float16*)alloc(COUT * COUT * 2);
  _Float16* qTp   = (_Float16*)alloc((size_t)NB * HW * CQK * 2);
  _Float16* kTp   = (_Float16*)alloc((size_t)NB * HW * CQK * 2);
  _Float16* vhp   = (_Float16*)alloc((size_t)NB * COUT * HW * 2);
  float*    energy= (float*)   alloc((size_t)NB * COUT * COUT * 4);
  _Float16* attnh = (_Float16*)alloc((size_t)NB * COUT * COUT * 2);
  (void)ws_size; (void)n_in; (void)in_sizes; (void)out_size;

  const int nx = NB * CIN * HW;
  const int nf = NB * COUT * HW;

  dam_cast_f32_f16<<<(nx + 255) / 256, 256, 0, stream>>>(x, xh, nx);
  dam_cast_f32_f16<<<8, 256, 0, stream>>>(q_w, qwh, CQK * COUT);
  dam_cast_f32_f16<<<8, 256, 0, stream>>>(k_w, kwh, CQK * COUT);
  dam_cast_f32_f16<<<64, 256, 0, stream>>>(v_w, vwh, COUT * COUT);
  dam_pack_conv_w<<<(COUT * KCONV + 255) / 256, 256, 0, stream>>>(
      conv_w, wA, COUT * KCONV);

  dam_conv_gemm<<<NB * 128, 256, 0, stream>>>(wA, xh, y);
  dam_bn_stats<<<COUT, 256, 0, stream>>>(y, bn_gamma, bn_beta, scale, shift);
  dam_bn_relu<<<(nf + 255) / 256, 256, 0, stream>>>(y, scale, shift, feat,
                                                    feath, nf);

  dam_proj_gemm<<<128, 256, 0, stream>>>(qwh, q_b, feath, qTp, 1, 1);
  dam_proj_gemm<<<128, 256, 0, stream>>>(kwh, k_b, feath, kTp, 1, 1);
  dam_proj_gemm<<<1024, 256, 0, stream>>>(vwh, v_b, feath, vhp, 8, 0);

  dam_cam_energy<<<32, 256, 0, stream>>>(feath, energy);
  dam_cam_attn<<<NB * COUT, COUT, 0, stream>>>(energy, attnh);
  dam_cam_out<<<1024, 256, 0, stream>>>(attnh, feath, feat, gca, out);

  dam_pam_fused<<<NB * 128, 256, 0, stream>>>(qTp, kTp, vhp, gpa, out);
}
